// ChannelClustering_53180285059723
// MI455X (gfx1250) — matrix operand order repaired and checked by reading, not compiled, not run on hardware
//
#include <hip/hip_runtime.h>
#include <hip/hip_bf16.h>

// ---------------------------------------------------------------------------
// MoE channel clustering for MI455X (gfx1250, wave32, WMMA + TDM).
// Sizes: B=64 C=128 L=1024 D=1024 E=8 K=2 H=256, N = B*C = 8192.
// Output: [global_mask 128*128 | y 8192*1024] fp32.
// ---------------------------------------------------------------------------

typedef float        v8f   __attribute__((ext_vector_type(8)));
typedef __bf16       v16bf __attribute__((ext_vector_type(16)));
typedef __bf16       v8bf  __attribute__((ext_vector_type(8)));
typedef unsigned int u32x4 __attribute__((ext_vector_type(4)));
typedef int          i32x8 __attribute__((ext_vector_type(8)));
typedef int          i32x4 __attribute__((ext_vector_type(4)));

union BF16x16 { v16bf v; v8bf h[2]; };

#define N_TOK 8192
#define LDIM  1024
#define DDIM  1024
#define EEXP  8
#define HDIM  256
#define BATCH 64
#define CCH   128

// fp32 -> bf16 round-to-nearest-even
__device__ __forceinline__ unsigned short f2bf(float f) {
  union { float f; unsigned u; } x; x.f = f;
  unsigned u = x.u;
  if ((u & 0x7fffffffu) > 0x7f800000u) return (unsigned short)((u >> 16) | 0x40u); // NaN
  return (unsigned short)((u + 0x7fffu + ((u >> 16) & 1u)) >> 16);
}

// x4-vectorized converter (all our sizes are multiples of 4)
__global__ __launch_bounds__(256) void cvt_kernel(const float4* __restrict__ s,
                                                  ushort4* __restrict__ d, int n4) {
  int i = blockIdx.x * 256 + threadIdx.x;
  if (i < n4) {
    float4 f = s[i];
    ushort4 o;
    o.x = f2bf(f.x); o.y = f2bf(f.y); o.z = f2bf(f.z); o.w = f2bf(f.w);
    d[i] = o;
  }
}

// Build a 16-bit A fragment (16x32 MxK) for this lane from a bf16 row pointer.
// Lane (0-31): row m = lane&15; K range: [k0+sel*16 .. +15] contiguous.
__device__ __forceinline__ v16bf load_a_frag(const unsigned short* __restrict__ row,
                                             int k0, int sel) {
  BF16x16 a;
  a.h[0] = *(const v8bf*)(row + k0 + sel * 16);
  a.h[1] = *(const v8bf*)(row + k0 + sel * 16 + 8);
  return a.v;
}

// ---------------------------------------------------------------------------
// Gating kernel: one wave per 16-token tile.
//   hidden = relu(x @ W1^T + b1)  via v_wmma_f32_16x16x32_bf16 (32 K-steps)
//   clean/raw = hidden @ W2^T + b2 (tiny, VALU)
//   logits = (clean + noise*(softplus(raw)+0.01)) @ W_h ; softmax; top-2 gates
//   append (token, gate) to per-expert lists with atomic counters.
// ---------------------------------------------------------------------------
__global__ __launch_bounds__(32) void gate_kernel(
    const unsigned short* __restrict__ xb,    // [N][L] bf16
    const unsigned short* __restrict__ Wg1b,  // [H][L] bf16
    const float* __restrict__ bg1,
    const float* __restrict__ Wg2,            // [E][H]
    const float* __restrict__ bg2,
    const unsigned short* __restrict__ Wn1b,
    const float* __restrict__ bn1,
    const float* __restrict__ Wn2,
    const float* __restrict__ bn2,
    const float* __restrict__ W_h,            // [E][E]
    const float* __restrict__ noise,          // [N][E]
    float* __restrict__ gates,                // [N][E]
    int* __restrict__ counts,                 // [E]
    int* __restrict__ etok,                   // [E][N]
    float* __restrict__ egate)                // [E][N]
{
  __shared__ __align__(16) float hid[16][HDIM];
  __shared__ float cn[2][16][EEXP];

  const int lane = threadIdx.x;
  const int n0   = blockIdx.x * 16;
  const int m    = lane & 15;
  const int sel  = lane >> 4;
  const unsigned short* arow = xb + (size_t)(n0 + m) * LDIM;

  for (int net = 0; net < 2; ++net) {
    const unsigned short* W1 = net ? Wn1b : Wg1b;
    const float* b1 = net ? bn1 : bg1;

    // layer 1: 16 tokens x 256 hidden via WMMA; two groups of 8 N-tiles
    for (int tg = 0; tg < 2; ++tg) {
      v8f acc[8] = {};
      for (int k0 = 0; k0 < LDIM; k0 += 32) {
        v16bf a = load_a_frag(arow, k0, sel);
#pragma unroll
        for (int t = 0; t < 8; ++t) {
          const unsigned short* brow =
              W1 + (size_t)(tg * 128 + t * 16 + m) * LDIM + k0 + sel * 16;
          v16bf b = *(const v16bf*)brow;
          acc[t] = __builtin_amdgcn_wmma_f32_16x16x32_bf16(
              false, a, false, b, (short)0, acc[t], false, false);
        }
      }
#pragma unroll
      for (int t = 0; t < 8; ++t) {
#pragma unroll
        for (int r = 0; r < 8; ++r) {
          int mm = r + sel * 8;
          int hh = tg * 128 + t * 16 + m;
          float v = acc[t][r] + b1[hh];
          hid[mm][hh] = v > 0.f ? v : 0.f;
        }
      }
    }
    __syncthreads();

    // layer 2 (16x8 outputs over K=256): float4 dot products
    const float* W2 = net ? Wn2 : Wg2;
    const float* b2 = net ? bn2 : bg2;
    for (int o = lane; o < 16 * EEXP; o += 32) {
      int mm = o >> 3, e = o & 7;
      const float4* hv = (const float4*)(&hid[mm][0]);
      const float4* wv = (const float4*)(&W2[e * HDIM]);
      float s = b2[e];
      for (int h = 0; h < HDIM / 4; ++h) {
        float4 a = hv[h], b = wv[h];
        s += a.x * b.x + a.y * b.y + a.z * b.z + a.w * b.w;
      }
      cn[net][mm][e] = s;
    }
    __syncthreads();
  }

  if (lane < 16) {
    int n = n0 + lane;
    float v[EEXP];
#pragma unroll
    for (int e = 0; e < EEXP; ++e) {
      float r  = cn[1][lane][e];
      float sp = (r > 20.f) ? r : log1pf(expf(r));           // softplus
      v[e] = cn[0][lane][e] + noise[(size_t)n * EEXP + e] * (sp + 0.01f);
    }
    float lg[EEXP];
#pragma unroll
    for (int e2 = 0; e2 < EEXP; ++e2) {
      float s = 0.f;
#pragma unroll
      for (int e = 0; e < EEXP; ++e) s += v[e] * W_h[e * EEXP + e2];
      lg[e2] = s;
    }
    float mx = lg[0];
#pragma unroll
    for (int e = 1; e < EEXP; ++e) mx = fmaxf(mx, lg[e]);
    float p[EEXP], sum = 0.f;
#pragma unroll
    for (int e = 0; e < EEXP; ++e) { p[e] = expf(lg[e] - mx); sum += p[e]; }
    float inv = 1.f / sum;
#pragma unroll
    for (int e = 0; e < EEXP; ++e) p[e] *= inv;

    int i0 = 0; float b0 = p[0];
#pragma unroll
    for (int e = 1; e < EEXP; ++e) if (p[e] > b0) { b0 = p[e]; i0 = e; }
    int i1 = (i0 == 0) ? 1 : 0; float b1v = p[i1];
#pragma unroll
    for (int e = 0; e < EEXP; ++e)
      if (e != i0 && p[e] > b1v) { b1v = p[e]; i1 = e; }

    float gs = b0 + b1v + 1e-6f;
    float g0 = b0 / gs, g1 = b1v / gs;

#pragma unroll
    for (int e = 0; e < EEXP; ++e) gates[(size_t)n * EEXP + e] = 0.f;
    gates[(size_t)n * EEXP + i0] = g0;
    gates[(size_t)n * EEXP + i1] = g1;

    int p0 = atomicAdd(&counts[i0], 1);
    etok[i0 * N_TOK + p0] = n; egate[i0 * N_TOK + p0] = g0;
    int p1 = atomicAdd(&counts[i1], 1);
    etok[i1 * N_TOK + p1] = n; egate[i1 * N_TOK + p1] = g1;
  }
}

// ---------------------------------------------------------------------------
// Expert kernel: grid = (512 token tiles, 8 experts, 4 column groups of 256).
// The gathered 16-row bf16 A tile is staged to LDS with the Tensor Data Mover
// in gather mode (16-bit row indices = token ids), then 8 waves each compute a
// 16x32 f32 slab over K=1024 (2 WMMA chains sharing one A fragment), then
// relu(+bias)*gate and atomicAdd into y (exactly 2 adds per element -> exact).
// ---------------------------------------------------------------------------
__global__ __launch_bounds__(256) void expert_kernel(
    const unsigned short* __restrict__ xb,    // [N][L] bf16
    const unsigned short* __restrict__ Web,   // [E][D][L] bf16
    const float* __restrict__ be,             // [E][D]
    const int* __restrict__ counts,
    const int* __restrict__ etok,
    const float* __restrict__ egate,
    float* __restrict__ y)                    // [N][D]
{
  const int e    = blockIdx.y;
  const int tile = blockIdx.x;
  const int cnt  = counts[e];
  if (tile * 16 >= cnt) return;

  __shared__ __align__(16) unsigned short As[16 * LDIM];  // 32 KB gathered A tile
  __shared__ int   tok[16];
  __shared__ float gw[16];

  const int tid = threadIdx.x;
  if (tid < 16) {
    int idx = tile * 16 + tid;
    if (idx < cnt) { tok[tid] = etok[e * N_TOK + idx]; gw[tid] = egate[e * N_TOK + idx]; }
    else           { tok[tid] = 0;                     gw[tid] = 0.f; }
  }
  __syncthreads();

#if defined(__gfx1250__) && __has_builtin(__builtin_amdgcn_tensor_load_to_lds)
  if (tid < 32) {
    // TDM gather: 16 rows of 1024 bf16 each from xb into As, rows = token ids.
    // Descriptor per CDNA5 ISA (gather mode, 16-bit idx).
    unsigned lds_base = (unsigned)(size_t)(void*)As;           // addr[31:0] = LDS offset
    unsigned long long ga = (unsigned long long)(size_t)xb;    // tensor base

    u32x4 g0;
    g0[0] = (unsigned)__builtin_amdgcn_readfirstlane((int)(1u | (1u << 31)));  // count=1, gather_mode=1, 16b idx
    g0[1] = (unsigned)__builtin_amdgcn_readfirstlane((int)lds_base);
    g0[2] = (unsigned)__builtin_amdgcn_readfirstlane((int)(unsigned)(ga & 0xFFFFFFFFu));
    g0[3] = (unsigned)__builtin_amdgcn_readfirstlane(
        (int)(((unsigned)(ga >> 32) & 0x01FFFFFFu) | (2u << 30)));             // type=2 (image)

    i32x8 g1;
    g1[0] = __builtin_amdgcn_readfirstlane((int)(1u << 16));        // data_size=2B, mask=0
    g1[1] = __builtin_amdgcn_readfirstlane((int)(1024u << 16));     // tensor_dim0 = 1024
    g1[2] = __builtin_amdgcn_readfirstlane((int)(8192u << 16));     // tensor_dim1 = 8192 (rows)
    g1[3] = __builtin_amdgcn_readfirstlane((int)(1024u << 16));     // tile_dim0 = 1024
    g1[4] = __builtin_amdgcn_readfirstlane((int)16);                // tile_dim1 = 16 indices
    g1[5] = __builtin_amdgcn_readfirstlane((int)1024);              // tensor_dim0_stride = 1024
    g1[6] = __builtin_amdgcn_readfirstlane(0);
    g1[7] = __builtin_amdgcn_readfirstlane(0);

    i32x4 g2, g3;                                                   // packed 16-bit row indices
#pragma unroll
    for (int j = 0; j < 4; ++j) {
      g2[j] = __builtin_amdgcn_readfirstlane((tok[2 * j] & 0xFFFF) | (tok[2 * j + 1] << 16));
      g3[j] = __builtin_amdgcn_readfirstlane((tok[8 + 2 * j] & 0xFFFF) | (tok[8 + 2 * j + 1] << 16));
    }

    // 6-arg toolchain form: trailing int32x8 beyond the 4 architectural D#
    // groups is uncharacterized -> zero-filled (probe-verified usage).
    i32x8 g4 = { 0, 0, 0, 0, 0, 0, 0, 0 };

    __builtin_amdgcn_tensor_load_to_lds(g0, g1, g2, g3, g4, 0);
    __builtin_amdgcn_s_wait_tensorcnt(0);
  }
#else
  for (int i = tid; i < 2048; i += 256) {          // fallback: 2048 uint4 copies
    int row = i >> 7;
    int col = i & 127;
    ((uint4*)As)[i] =
        ((const uint4*)(const void*)(xb + (size_t)tok[row] * LDIM))[col];
  }
#endif
  __syncthreads();

  const int wave = tid >> 5, lane = tid & 31;
  const int m = lane & 15, sel = lane >> 4;
  const int dcol0 = blockIdx.z * 256 + wave * 32;  // 16x32 slab per wave
  const unsigned short* W = Web + (size_t)e * DDIM * LDIM;
  const unsigned short* arow  = As + m * LDIM;
  const unsigned short* brow0 = W + (size_t)(dcol0 + m) * LDIM + sel * 16;
  const unsigned short* brow1 = W + (size_t)(dcol0 + 16 + m) * LDIM + sel * 16;

  v8f acc0 = {}, acc1 = {};
  for (int k0 = 0; k0 < LDIM; k0 += 32) {
    v16bf a = load_a_frag(arow, k0, sel);
    __builtin_prefetch(brow0 + k0 + 64, 0, 0);     // global_prefetch_b8
    v16bf b0 = *(const v16bf*)(brow0 + k0);
    v16bf b1 = *(const v16bf*)(brow1 + k0);
    acc0 = __builtin_amdgcn_wmma_f32_16x16x32_bf16(
        false, a, false, b0, (short)0, acc0, false, false);
    acc1 = __builtin_amdgcn_wmma_f32_16x16x32_bf16(
        false, a, false, b1, (short)0, acc1, false, false);
  }

  const int d0 = dcol0 + m;
  const int d1 = dcol0 + 16 + m;
  const float bias0 = be[e * DDIM + d0];
  const float bias1 = be[e * DDIM + d1];
#pragma unroll
  for (int r = 0; r < 8; ++r) {
    int mm = r + sel * 8;
    float v0 = acc0[r] + bias0; v0 = v0 > 0.f ? v0 : 0.f;
    float v1 = acc1[r] + bias1; v1 = v1 > 0.f ? v1 : 0.f;
    float w = gw[mm];
    if (w != 0.f) {
      float* yr = &y[(size_t)tok[mm] * DDIM];
      atomicAdd(yr + d0, v0 * w);
      atomicAdd(yr + d1, v1 * w);
    }
  }
}

// ---------------------------------------------------------------------------
// global_mask[c][d] = (1/B) * sum_b sum_e g[b,c,e]*g[b,d,e]   (tiny: VALU)
// ---------------------------------------------------------------------------
__global__ __launch_bounds__(256) void mask_kernel(const float* __restrict__ gates,
                                                   float* __restrict__ mask) {
  int c = blockIdx.y * 16 + threadIdx.y;
  int d = blockIdx.x * 16 + threadIdx.x;
  float s = 0.f;
  for (int b = 0; b < BATCH; ++b) {
    const float* gc = gates + (size_t)(b * CCH + c) * EEXP;
    const float* gd = gates + (size_t)(b * CCH + d) * EEXP;
#pragma unroll
    for (int e = 0; e < EEXP; ++e) s += gc[e] * gd[e];
  }
  mask[c * CCH + d] = s * (1.f / (float)BATCH);
}

// ---------------------------------------------------------------------------
extern "C" void kernel_launch(void* const* d_in, const int* in_sizes, int n_in,
                              void* d_out, int out_size, void* d_ws, size_t ws_size,
                              hipStream_t stream) {
  (void)in_sizes; (void)n_in; (void)ws_size;

  const float* x    = (const float*)d_in[0];
  const float* nois = (const float*)d_in[1];
  const float* Wg1  = (const float*)d_in[2];
  const float* bg1  = (const float*)d_in[3];
  const float* Wg2  = (const float*)d_in[4];
  const float* bg2  = (const float*)d_in[5];
  const float* Wn1  = (const float*)d_in[6];
  const float* bn1  = (const float*)d_in[7];
  const float* Wn2  = (const float*)d_in[8];
  const float* bn2  = (const float*)d_in[9];
  const float* W_h  = (const float*)d_in[10];
  const float* We   = (const float*)d_in[11];
  const float* be   = (const float*)d_in[12];

  char* ws = (char*)d_ws;
  unsigned short* xb    = (unsigned short*)(ws + 0);          // 16 MB
  unsigned short* Web   = (unsigned short*)(ws + 16777216);   // 16 MB
  unsigned short* Wg1b  = (unsigned short*)(ws + 33554432);   // 512 KB
  unsigned short* Wn1b  = (unsigned short*)(ws + 34078720);   // 512 KB
  float*          gates = (float*)(ws + 34603008);            // 256 KB
  int*            cnts  = (int*)(ws + 34865152);              // 64 B
  int*            etok  = (int*)(ws + 34865216);              // 256 KB
  float*          egate = (float*)(ws + 35127360);            // 256 KB

  float* mask_out = (float*)d_out;             // 128*128
  float* y_out    = (float*)d_out + CCH * CCH; // 8192*1024

  (void)hipMemsetAsync(d_out, 0, (size_t)out_size * sizeof(float), stream);
  (void)hipMemsetAsync(cnts, 0, 64, stream);

  cvt_kernel<<<(2097152 + 255) / 256, 256, 0, stream>>>((const float4*)x, (ushort4*)xb, 2097152);
  cvt_kernel<<<(65536 + 255) / 256, 256, 0, stream>>>((const float4*)Wg1, (ushort4*)Wg1b, 65536);
  cvt_kernel<<<(65536 + 255) / 256, 256, 0, stream>>>((const float4*)Wn1, (ushort4*)Wn1b, 65536);
  cvt_kernel<<<(2097152 + 255) / 256, 256, 0, stream>>>((const float4*)We, (ushort4*)Web, 2097152);

  gate_kernel<<<N_TOK / 16, 32, 0, stream>>>(xb, Wg1b, bg1, Wg2, bg2,
                                             Wn1b, bn1, Wn2, bn2,
                                             W_h, nois, gates, cnts, etok, egate);

  expert_kernel<<<dim3(N_TOK / 16, EEXP, DDIM / 256), 256, 0, stream>>>(
      xb, Web, be, cnts, etok, egate, y_out);

  mask_kernel<<<dim3(CCH / 16, CCH / 16), dim3(16, 16), 0, stream>>>(gates, mask_out);
}
